// DLKAConvBlock_11020886082079
// MI455X (gfx1250) — matrix operand order repaired and checked by reading, not compiled, not run on hardware
//
#include <hip/hip_runtime.h>
#include <hip/hip_bf16.h>
#include <math.h>

// ---------------------------------------------------------------------------
// DLKA conv block for MI455X (gfx1250): bf16 WMMA implicit-GEMM convolutions
// with double-buffered LDS B-tiles (global-load latency hidden under WMMA),
// + VALU bilinear-gather deformable depthwise passes.
// ---------------------------------------------------------------------------

typedef __bf16 bf16;
typedef __attribute__((ext_vector_type(16))) __bf16 v16bf;
typedef __attribute__((ext_vector_type(8)))  __bf16 v8bf;
typedef __attribute__((ext_vector_type(2)))  __bf16 v2bf;
typedef __attribute__((ext_vector_type(8)))  float  v8f;

#define NB   4
#define NC   64
#define IMH  256
#define IMW  256
#define HWP  (IMH * IMW)   // 65536
#define SBS  80            // LDS B-tile row stride in bf16 (64 + 16 pad, keeps 32B align)

__device__ __forceinline__ bf16 f2bf(float f) { return (bf16)f; }

// ---------------------------------------------------------------------------
// Weight packing: f32 [CoutReal][Kreal] (contiguous) -> bf16 [CoutPad][Kreal]
// ---------------------------------------------------------------------------
__global__ void k_pack_weights(const float* __restrict__ w, bf16* __restrict__ wp,
                               int coutReal, int kReal) {
    const int co = blockIdx.x;          // 0..coutPad-1
    for (int k = threadIdx.x; k < kReal; k += blockDim.x) {
        float v = (co < coutReal) ? w[(size_t)co * kReal + k] : 0.0f;
        wp[(size_t)co * kReal + k] = f2bf(v);
    }
}

// ---------------------------------------------------------------------------
// Implicit-GEMM convolution with WMMA bf16, software-pipelined.
// Block tile: 64 pixels (one row segment) x CTILES*16 couts, 64 K per stage.
// Wave tile: 16 couts x 64 pixels = 4 v8f accumulators; A fragment reused 4x.
// Pipeline per iteration: fetch(next K-chunk -> regs) | 8x WMMA(cur chunk)
//                         | commit(regs -> other LDS buffer) | one barrier.
// ---------------------------------------------------------------------------
#define MODE_NONE           0
#define MODE_BIAS           1
#define MODE_BIAS_GELU      2
#define MODE_BIAS_MUL       3   // (acc+bias) * aux       (g1 conv fused with t=u*a)
#define MODE_BIAS_RES_LRELU 4   // leaky(acc+bias+aux)    (p2 conv fused w/ residual)

template <int KH, int KW, int DIL, int PAD, int CTILES, int MODE>
__global__ __launch_bounds__(CTILES * 32)
void k_conv_wmma(const float* __restrict__ in, const bf16* __restrict__ wp,
                 const float* __restrict__ bias, const float* __restrict__ aux,
                 float* __restrict__ out, int Cin, int CoutReal, int Kpad) {
    constexpr int NT  = CTILES * 32;                 // threads per block
    constexpr int EPT = (2048 + NT - 1) / NT;        // bf16 pairs staged per thread

    const int x0   = blockIdx.x * 64;
    const int y    = blockIdx.y;
    const int b    = blockIdx.z;
    const int tid  = threadIdx.x;
    const int lane = tid & 31;
    const int wave = tid >> 5;
    const int n    = lane & 15;   // pixel-in-subtile / A row
    const int half = lane >> 4;   // lane-group selector per ISA striping

    // Double-buffered B tile: [buf][pixel][k] bf16, 64 pix x 64 K (+16 pad/row)
    __shared__ __align__(64) bf16 sB[2][64 * SBS];

    const int  Kreal = Cin * KH * KW;       // always a multiple of 64 here
    const int  cout0 = wave * 16;
    const bf16* aBase = wp + (size_t)(cout0 + n) * Kpad;

    v8f  acc[4] = {};
    v2bf stage[EPT];

    // ---- im2col fetch of one 64K x 64pix chunk into registers -------------
    auto fetch = [&](int kc) {
#pragma unroll
        for (int i = 0; i < EPT; ++i) {
            const int e = tid + i * NT;
            if (e < 2048) {
                const int pix = e & 63;      // consecutive lanes -> coalesced x
                const int kp  = e >> 6;      // K pair index 0..31
                v2bf pr;
#pragma unroll
                for (int j = 0; j < 2; ++j) {
                    const int k   = kc + kp * 2 + j;
                    const int cin = k / (KH * KW);
                    const int rem = k - cin * (KH * KW);
                    const int ki  = rem / KW;
                    const int kj  = rem - ki * KW;
                    const int yy  = y + DIL * ki - PAD;
                    const int xx  = x0 + pix + DIL * kj - PAD;
                    float v = 0.0f;
                    if (yy >= 0 && yy < IMH && xx >= 0 && xx < IMW)
                        v = in[(((size_t)b * Cin + cin) * IMH + yy) * IMW + xx];
                    pr[j] = f2bf(v);
                }
                stage[i] = pr;
            }
        }
    };
    // ---- commit staged registers into LDS buffer --------------------------
    auto commit = [&](int buf) {
#pragma unroll
        for (int i = 0; i < EPT; ++i) {
            const int e = tid + i * NT;
            if (e < 2048) {
                const int pix = e & 63;
                const int kp  = e >> 6;
                *(v2bf*)(&sB[buf][pix * SBS + kp * 2]) = stage[i];
            }
        }
    };

    // ---- prologue: stage chunk 0 ------------------------------------------
    fetch(0);
    commit(0);
    __syncthreads();

    for (int kc = 0; kc < Kreal; kc += 64) {
        const int cur = (kc >> 6) & 1;
        const bool more = (kc + 64) < Kreal;

        if (more) fetch(kc + 64);          // global loads in flight during WMMA

        // -------- 2 K-halves x 4 pixel sub-tiles = 8 WMMA ------------------
#pragma unroll
        for (int h = 0; h < 2; ++h) {
            // A fragment: lanes 0-15 K=+0..15, lanes 16-31 K=+16..31
            const bf16* ap = aBase + kc + h * 32 + half * 16;
            v8bf alo = *(const v8bf*)(ap);
            v8bf ahi = *(const v8bf*)(ap + 8);
            v16bf afrag;
#pragma unroll
            for (int i = 0; i < 8; ++i) { afrag[i] = alo[i]; afrag[8 + i] = ahi[i]; }
#pragma unroll
            for (int p = 0; p < 4; ++p) {
                // B fragment: lanes 0-15 K lo-half of pixel, lanes 16-31 K hi-half
                v16bf bfrag = *(const v16bf*)(&sB[cur][(p * 16 + n) * SBS + h * 32 + half * 16]);
                acc[p] = __builtin_amdgcn_wmma_f32_16x16x32_bf16(
                    false, afrag, false, bfrag, (short)0, acc[p], false, false);
            }
        }

        if (more) commit(cur ^ 1);         // DS stores after matrix issue
        __syncthreads();                   // single barrier per K-chunk
    }

    // -------- epilogue / writeout -----------------------------------------
#pragma unroll
    for (int p = 0; p < 4; ++p) {
        const size_t pix = (size_t)y * IMW + x0 + p * 16 + n;
#pragma unroll
        for (int r = 0; r < 8; ++r) {
            const int co = cout0 + half * 8 + r;
            if (co < CoutReal) {
                float v = acc[p][r];
                const size_t idx = ((size_t)b * CoutReal + co) * HWP + pix;
                if (MODE == MODE_NONE) {
                    out[idx] = v;
                } else if (MODE == MODE_BIAS) {
                    out[idx] = v + bias[co];
                } else if (MODE == MODE_BIAS_GELU) {
                    const float z = v + bias[co];
                    out[idx] = 0.5f * z * (1.0f + erff(z * 0.70710678118654752f));
                } else if (MODE == MODE_BIAS_MUL) {
                    out[idx] = (v + bias[co]) * aux[idx];
                } else { // MODE_BIAS_RES_LRELU
                    const float z = v + bias[co] + aux[idx];
                    out[idx] = z >= 0.0f ? z : 0.2f * z;
                }
            }
        }
    }
}

// ---------------------------------------------------------------------------
// InstanceNorm: per-(b,c) stats over 65536 pixels, then elementwise apply.
// ---------------------------------------------------------------------------
__global__ __launch_bounds__(256)
void k_instnorm_stats(const float* __restrict__ x, float* __restrict__ mean,
                      float* __restrict__ rstd) {
    const int bc = blockIdx.x;
    const float* p = x + (size_t)bc * HWP;
    float s = 0.0f, s2 = 0.0f;
    for (int i = threadIdx.x; i < HWP; i += 256) {
        const float v = p[i];
        s += v; s2 += v * v;
    }
    __shared__ float sh[512];
    sh[threadIdx.x] = s;
    sh[256 + threadIdx.x] = s2;
    __syncthreads();
    for (int off = 128; off > 0; off >>= 1) {
        if (threadIdx.x < off) {
            sh[threadIdx.x]       += sh[threadIdx.x + off];
            sh[256 + threadIdx.x] += sh[256 + threadIdx.x + off];
        }
        __syncthreads();
    }
    if (threadIdx.x == 0) {
        const float m = sh[0] * (1.0f / HWP);
        const float v = sh[256] * (1.0f / HWP) - m * m;
        mean[bc] = m;
        rstd[bc] = rsqrtf(v + 1e-5f);
    }
}

__global__ __launch_bounds__(256)
void k_instnorm_apply(const float* __restrict__ x, const float* __restrict__ mean,
                      const float* __restrict__ rstd, float* __restrict__ out) {
    const size_t i = (size_t)blockIdx.x * 256 + threadIdx.x;
    const int bc = (int)(i >> 16);      // HWP == 1<<16
    out[i] = (x[i] - mean[bc]) * rstd[bc];
}

// ---------------------------------------------------------------------------
// Deformable depthwise conv (one offset group), bilinear sampling with zero
// padding. Block = one (b,c,y) row, thread = x. Offset reads coalesced along
// x; depthwise weights are wave-uniform -> scalar loads.
// ---------------------------------------------------------------------------
__device__ __forceinline__ float dsamp(const float* __restrict__ p, int yy, int xx) {
    if (yy < 0 || yy >= IMH || xx < 0 || xx >= IMW) return 0.0f;
    return p[yy * IMW + xx];
}

template <int KH, int KW, int DIL, int PAD>
__global__ __launch_bounds__(256)
void k_deform_dw(const float* __restrict__ in, const float* __restrict__ off,
                 const float* __restrict__ w, float* __restrict__ out) {
    const int K = KH * KW;
    const int x = threadIdx.x;
    const int y = blockIdx.x;
    const int c = blockIdx.y;
    const int b = blockIdx.z;

    const float* inp  = in  + ((size_t)b * NC + c) * HWP;
    const float* offp = off + ((size_t)b * (2 * K)) * HWP + (size_t)y * IMW + x;

    float acc = 0.0f;
#pragma unroll 1
    for (int k = 0; k < K; ++k) {
        const int   ki = k / KW;
        const int   kj = k - ki * KW;
        const float wv = w[c * K + k];                 // uniform -> scalar load
        const float dy = offp[(size_t)(2 * k)     * HWP];
        const float dx = offp[(size_t)(2 * k + 1) * HWP];
        const float py = (float)(y - PAD + DIL * ki) + dy;
        const float px = (float)(x - PAD + DIL * kj) + dx;
        const float fy = floorf(py), fx = floorf(px);
        const float ty = py - fy,    tx = px - fx;
        const int iy0 = (int)fy, ix0 = (int)fx;
        const float v00 = dsamp(inp, iy0,     ix0);
        const float v01 = dsamp(inp, iy0,     ix0 + 1);
        const float v10 = dsamp(inp, iy0 + 1, ix0);
        const float v11 = dsamp(inp, iy0 + 1, ix0 + 1);
        const float top = v00 + (v01 - v00) * tx;
        const float bot = v10 + (v11 - v10) * tx;
        acc += (top + (bot - top) * ty) * wv;
    }
    out[((size_t)b * NC + c) * HWP + (size_t)y * IMW + x] = acc;
}

// ---------------------------------------------------------------------------
// Launch sequence
// ---------------------------------------------------------------------------
extern "C" void kernel_launch(void* const* d_in, const int* in_sizes, int n_in,
                              void* d_out, int out_size, void* d_ws, size_t ws_size,
                              hipStream_t stream) {
    (void)in_sizes; (void)n_in; (void)out_size; (void)ws_size;

    const float* x      = (const float*)d_in[0];
    const float* conv_w = (const float*)d_in[1];
    const float* p1_w   = (const float*)d_in[2];
    const float* p1_b   = (const float*)d_in[3];
    const float* off0_w = (const float*)d_in[4];
    const float* off0_b = (const float*)d_in[5];
    const float* dw0_w  = (const float*)d_in[6];
    const float* offs_w = (const float*)d_in[7];
    const float* offs_b = (const float*)d_in[8];
    const float* dws_w  = (const float*)d_in[9];
    const float* g1_w   = (const float*)d_in[10];
    const float* g1_b   = (const float*)d_in[11];
    const float* p2_w   = (const float*)d_in[12];
    const float* p2_b   = (const float*)d_in[13];
    float* out = (float*)d_out;

    // ---------------- workspace layout (f32, then bf16 packed weights) -----
    const size_t fT = (size_t)NB * NC * HWP;        // 16,777,216 floats
    float* buf0    = (float*)d_ws;                  // conv3x3 raw -> reused as deform2 out
    float* buf1    = buf0 + fT;                     // h_norm (shortcut)
    float* buf2    = buf1 + fT;                     // t = gelu(p1(h))  (== u)
    float* buf4    = buf2 + fT;                     // deform1 out -> reused as u*a
    float* off0buf = buf4 + fT;                     // [B,50,H,W]
    const size_t off0E = (size_t)NB * 50 * HWP;
    float* offsbuf = off0buf + off0E;               // [B,98,H,W]
    const size_t offsE = (size_t)NB * 98 * HWP;
    float* meanb = offsbuf + offsE;                 // [B*C]
    float* rstdb = meanb + NB * NC;                 // [B*C]
    bf16* wp_conv3 = (bf16*)(rstdb + NB * NC);      // 64  x 576
    bf16* wp_p1    = wp_conv3 + 64 * 576;           // 64  x 64
    bf16* wp_off0  = wp_p1    + 64 * 64;            // 64  x 1600 (50 padded to 64)
    bf16* wp_offs  = wp_off0  + 64 * 1600;          // 112 x 3136 (98 padded to 112)
    bf16* wp_g1    = wp_offs  + (size_t)112 * 3136; // 64  x 64
    bf16* wp_p2    = wp_g1    + 64 * 64;            // 64  x 64

    // ---------------- pack weights to bf16 [CoutPad][K] ---------------------
    hipLaunchKernelGGL(k_pack_weights, dim3(64),  dim3(256), 0, stream, conv_w, wp_conv3, 64, 576);
    hipLaunchKernelGGL(k_pack_weights, dim3(64),  dim3(256), 0, stream, p1_w,   wp_p1,   64, 64);
    hipLaunchKernelGGL(k_pack_weights, dim3(64),  dim3(256), 0, stream, off0_w, wp_off0, 50, 1600);
    hipLaunchKernelGGL(k_pack_weights, dim3(112), dim3(256), 0, stream, offs_w, wp_offs, 98, 3136);
    hipLaunchKernelGGL(k_pack_weights, dim3(64),  dim3(256), 0, stream, g1_w,   wp_g1,   64, 64);
    hipLaunchKernelGGL(k_pack_weights, dim3(64),  dim3(256), 0, stream, p2_w,   wp_p2,   64, 64);

    const dim3 convGrid(IMW / 64, IMH, NB);

    // 1) conv3x3 (no bias) -> buf0
    hipLaunchKernelGGL((k_conv_wmma<3,3,1,1,4,MODE_NONE>), convGrid, dim3(4*32), 0, stream,
                       x, wp_conv3, nullptr, nullptr, buf0, NC, NC, 576);
    // 2) instance norm -> buf1 (shortcut)
    hipLaunchKernelGGL(k_instnorm_stats, dim3(NB*NC), dim3(256), 0, stream, buf0, meanb, rstdb);
    hipLaunchKernelGGL(k_instnorm_apply, dim3((unsigned)(fT/256)), dim3(256), 0, stream,
                       buf0, meanb, rstdb, buf1);
    // 3) t = gelu(p1(h)) -> buf2  (== u)
    hipLaunchKernelGGL((k_conv_wmma<1,1,1,0,4,MODE_BIAS_GELU>), convGrid, dim3(4*32), 0, stream,
                       buf1, wp_p1, p1_b, nullptr, buf2, NC, NC, 64);
    // 4) off0 = conv5x5(t) + b -> off0buf
    hipLaunchKernelGGL((k_conv_wmma<5,5,1,2,4,MODE_BIAS>), convGrid, dim3(4*32), 0, stream,
                       buf2, wp_off0, off0_b, nullptr, off0buf, NC, 50, 1600);
    // 5) a = deform_dw 5x5 -> buf4
    hipLaunchKernelGGL((k_deform_dw<5,5,1,2>), dim3(IMH, NC, NB), dim3(256), 0, stream,
                       buf2, off0buf, dw0_w, buf4);
    // 6) offs = conv7x7 dil3(a) + b -> offsbuf
    hipLaunchKernelGGL((k_conv_wmma<7,7,3,9,7,MODE_BIAS>), convGrid, dim3(7*32), 0, stream,
                       buf4, wp_offs, offs_b, nullptr, offsbuf, NC, 98, 3136);
    // 7) a = deform_dw 7x7 dil3 -> buf0 (reuse)
    hipLaunchKernelGGL((k_deform_dw<7,7,3,9>), dim3(IMH, NC, NB), dim3(256), 0, stream,
                       buf4, offsbuf, dws_w, buf0);
    // 8) t = u * (g1(a) + b) -> buf4 (reuse)  [fused multiply]
    hipLaunchKernelGGL((k_conv_wmma<1,1,1,0,4,MODE_BIAS_MUL>), convGrid, dim3(4*32), 0, stream,
                       buf0, wp_g1, g1_b, buf2, buf4, NC, NC, 64);
    // 9) out = leaky(p2(t) + b + shortcut) -> d_out  [fused residual+LeakyReLU]
    hipLaunchKernelGGL((k_conv_wmma<1,1,1,0,4,MODE_BIAS_RES_LRELU>), convGrid, dim3(4*32), 0, stream,
                       buf4, wp_p2, p2_b, buf1, out, NC, NC, 64);
}
